// Block_self_attention_inter_intra_3D_52596169507373
// MI455X (gfx1250) — matrix operand order repaired and checked by reading, not compiled, not run on hardware
//
#include <hip/hip_runtime.h>
#include <hip/hip_bf16.h>

typedef __attribute__((ext_vector_type(16))) __bf16 v16bf;
typedef __attribute__((ext_vector_type(8)))  float  v8f;

#define BATCH 2
#define CCH   64
#define HH    48
#define WW    48
#define TT    16
#define NPOS  (HH*WW*TT)   // 36864
#define NMAX  3456         // largest block: 24*24*6

union FragAB { uint4 q[2]; unsigned u[8]; v16bf v; };

__device__ __forceinline__ unsigned short f2bf(float f) {
    unsigned u = __float_as_uint(f);
    unsigned r = (u + 0x7FFFu + ((u >> 16) & 1u)) >> 16;
    return (unsigned short)r;
}

// pack two floats into packed bf16, round-to-nearest, one v_perm_b32
__device__ __forceinline__ unsigned pk2bf_rn(float lo, float hi) {
    unsigned ul = __float_as_uint(lo) + 0x8000u;
    unsigned uh = __float_as_uint(hi) + 0x8000u;
    // D bytes = [ul.b2, ul.b3, uh.b2, uh.b3]  (sel 0-3 -> src1, 4-7 -> src0)
    return __builtin_amdgcn_perm(uh, ul, 0x07060302u);
}

__device__ __forceinline__ v8f wmma_bf16(v16bf a, v16bf b, v8f c) {
    // D = A(16x32 bf16) * B(32x16 bf16) + C(16x16 f32)
    return __builtin_amdgcn_wmma_f32_16x16x32_bf16(false, a, false, b, (short)0, c,
                                                   false, false);
}

// ---------------------------------------------------------------------------
// Kernel 1: per-voxel 1x1x1 conv projections q/k/v = W*x + b (fp32 -> bf16).
// Q is pre-scaled by log2(e) so the attention kernel can use exp2 directly.
//   q_pm, k_pm : [b][n][c]  (position-major, channel-contiguous)
//   v_cm       : [b][c][n]  (channel-major)
// ---------------------------------------------------------------------------
__global__ void __launch_bounds__(256)
qkv_proj_kernel(const float* __restrict__ x,
                const float* __restrict__ Wq, const float* __restrict__ bq,
                const float* __restrict__ Wk, const float* __restrict__ bk,
                const float* __restrict__ Wv, const float* __restrict__ bv,
                unsigned short* __restrict__ q_pm,
                unsigned short* __restrict__ k_pm,
                unsigned short* __restrict__ v_cm) {
    const float LOG2E = 1.44269504088896340736f;
    int gid = blockIdx.x * 256 + threadIdx.x;   // 0 .. BATCH*NPOS-1
    int b = gid / NPOS;
    int n = gid - b * NPOS;

    float xr[CCH];
    #pragma unroll
    for (int ci = 0; ci < CCH; ++ci)
        xr[ci] = x[(size_t)(b * CCH + ci) * NPOS + n];

    unsigned* qout = (unsigned*)(q_pm + ((size_t)b * NPOS + n) * CCH);
    unsigned* kout = (unsigned*)(k_pm + ((size_t)b * NPOS + n) * CCH);

    for (int co = 0; co < CCH; co += 2) {
        float aq0 = bq[co],     ak0 = bk[co],     av0 = bv[co];
        float aq1 = bq[co + 1], ak1 = bk[co + 1], av1 = bv[co + 1];
        #pragma unroll
        for (int ci = 0; ci < CCH; ++ci) {
            float xv = xr[ci];
            aq0 += Wq[co * CCH + ci] * xv;
            ak0 += Wk[co * CCH + ci] * xv;
            av0 += Wv[co * CCH + ci] * xv;
            aq1 += Wq[(co + 1) * CCH + ci] * xv;
            ak1 += Wk[(co + 1) * CCH + ci] * xv;
            av1 += Wv[(co + 1) * CCH + ci] * xv;
        }
        aq0 *= LOG2E;  aq1 *= LOG2E;   // energy ends up pre-scaled for exp2
        qout[co >> 1] = pk2bf_rn(aq0, aq1);
        kout[co >> 1] = pk2bf_rn(ak0, ak1);
        v_cm[((size_t)(b * CCH + co)) * NPOS + n]     = f2bf(av0);
        v_cm[((size_t)(b * CCH + co + 1)) * NPOS + n] = f2bf(av1);
    }
}

// ---------------------------------------------------------------------------
// Kernel 2: block attention. 8 waves/WG work on the SAME block; K/V chunk
// tiles are staged into LDS via async-to-LDS ops (double-buffered, ASYNCcnt).
// K columns staged even/odd-interleaved so each lane's two P values per row
// are m-adjacent -> packed b32 P stores. Un-shifted softmax via native exp2
// (Q pre-scaled by log2 e); row normalizer via WMMA against all-ones operand.
// ---------------------------------------------------------------------------
__global__ void __launch_bounds__(256)
block_attn_kernel(const float* __restrict__ x,
                  const unsigned short* __restrict__ q_pm,
                  const unsigned short* __restrict__ k_pm,
                  const unsigned short* __restrict__ v_cm,
                  float* __restrict__ out) {
    __shared__ int mtab[NMAX];                   // block-local m -> spatial n
    __shared__ unsigned short Klds[2][32][64];   // [buf][col'][c]  4KB each
    __shared__ unsigned short Vlds[2][64][32];   // [buf][c][m]     4KB each
    __shared__ unsigned short Plds[8][16][32];   // per-wave P tile (bf16)

    int wg  = blockIdx.x;          // 0..575
    int b   = wg / 288;
    int rem = wg - b * 288;
    int blk = rem >> 3;            // 0..35
    int t8  = rem & 7;
    int bi = blk / 12;
    int bj = (blk / 4) % 3;
    int bz = blk & 3;

    int sx = bi * 16, sy = bj * 16, sz = bz * 4;
    int dy = (bj == 2) ? 16 : 24;
    int dz = (bz == 3) ? 4 : 6;
    int dx = (bi == 2) ? 16 : 24;
    int Nblk = dx * dy * dz;       // always a multiple of 32

    int tid = threadIdx.x;
    for (int m = tid; m < Nblk; m += 256) {
        int lz = m % dz;
        int r2 = m / dz;
        int ly = r2 % dy;
        int lx = r2 / dy;
        mtab[m] = ((sx + lx) * WW + (sy + ly)) * TT + (sz + lz);
    }
    __syncthreads();

    int wave = tid >> 5;
    int lane = tid & 31;
    int col  = lane & 15;          // B/D column, A row
    int hif  = lane >> 4;          // which 16-lane half

    int tile  = t8 * 8 + wave;     // 0..63: which 16 owned rows
    int rbase = hif * 8;           // D-layout rows for this lane

    // Global spatial n for this lane's 8 D-layout rows (owned region 16x16x4)
    int nrow[8];
    #pragma unroll
    for (int r = 0; r < 8; ++r) {
        int idx = tile * 16 + rbase + r;
        int oz = idx & 3, oy = (idx >> 2) & 15, ox = idx >> 6;
        nrow[r] = ((sx + ox) * WW + (sy + oy)) * TT + (sz + oz);
    }

    // ---- Q A-operand fragments (channels 0..31 and 32..63) ----
    int idxA = tile * 16 + col;
    int ozA = idxA & 3, oyA = (idxA >> 2) & 15, oxA = idxA >> 6;
    int nA  = ((sx + oxA) * WW + (sy + oyA)) * TT + (sz + ozA);
    const unsigned short* qb = q_pm + (size_t)b * NPOS * CCH;   // uniform base
    const unsigned short* qrow = qb + (unsigned)nA * CCH;
    int koff = hif * 16;           // A layout: lanes 0-15 K 0..15; 16-31: 16..31
    FragAB aq0, aq1;
    aq0.q[0] = *(const uint4*)(qrow + koff);
    aq0.q[1] = *(const uint4*)(qrow + koff + 8);
    aq1.q[0] = *(const uint4*)(qrow + koff + 32);
    aq1.q[1] = *(const uint4*)(qrow + koff + 40);

    int chOffK = hif * 16;         // B layout: lanes 0-15 hold K rows 0..15, 16-31 hold 16..31
    int kmoff  = hif * 16;         // A layout K offsets for the P operand (contiguous)
    int moff   = hif * 16;         // B layout m offsets for the V^T operand

    const unsigned short* kb = k_pm + (size_t)b * NPOS * CCH;   // uniform base
    const unsigned short* vb = v_cm + (size_t)b * CCH * NPOS;   // uniform base

    // All-ones B operand (bf16 1.0 pairs) for the row-sum WMMA
    FragAB bones;
    #pragma unroll
    for (int j = 0; j < 8; ++j) bones.u[j] = 0x3F803F80u;

    v8f Oacc[4];
    v8f Lacc = (v8f){};
    #pragma unroll
    for (int t = 0; t < 4; ++t) Oacc[t] = (v8f){};

    // ---- cooperative async K/V chunk staging (double-buffered) ----
    int sm  = tid >> 3;            // K staging: chunk-local m (0..31)
    int sc8 = (tid & 7) << 3;      // K staging: channel base
    // even m -> tile-0 columns 0..15, odd m -> tile-1 columns 16..31
    int krow = ((sm & 1) << 4) | (sm >> 1);
    auto stage = [&](int chunk) {
        int buf = chunk & 1;
        int mb  = chunk * 32;
        int nK  = mtab[mb + sm];
        unsigned klds  = (unsigned)(uintptr_t)&Klds[buf][krow][sc8];
        unsigned kgoff = ((unsigned)nK * CCH + (unsigned)sc8) * 2u;
        asm volatile("global_load_async_to_lds_b128 %0, %1, %2"
                     :: "v"(klds), "v"(kgoff), "s"(kb) : "memory");
        #pragma unroll
        for (int k = 0; k < 4; ++k) {
            int idx = tid + (k << 8);
            int c = idx & 63;
            int p = idx >> 6;                   // m-pair 0..15
            int nV = mtab[mb + 2 * p];          // m pairs are t-adjacent in memory
            unsigned vlds  = (unsigned)(uintptr_t)&Vlds[buf][c][2 * p];
            unsigned vgoff = ((unsigned)c * NPOS + (unsigned)nV) * 2u;
            asm volatile("global_load_async_to_lds_b32 %0, %1, %2"
                         :: "v"(vlds), "v"(vgoff), "s"(vb) : "memory");
        }
        // prefetch the chunk after this one (uniform condition: no EXEC churn)
        if (mb + 64 <= Nblk)
            __builtin_prefetch(kb + (unsigned)mtab[mb + 32 + sm] * CCH, 0, 1);
    };

    int nChunks = Nblk >> 5;
    stage(0);

    for (int ch = 0; ch < nChunks; ++ch) {
        // async writes to buf[ch&1] must land before any wave reads them
        asm volatile("s_wait_asynccnt 0x0" ::: "memory");
        __syncthreads();
        if (ch + 1 < nChunks) stage(ch + 1);
        int buf = ch & 1;

        // ---- K fragments from LDS (tile0 = even m, tile1 = odd m) ----
        const unsigned short* kr0 = &Klds[buf][col][0];
        const unsigned short* kr1 = &Klds[buf][col + 16][0];
        FragAB bk00, bk01, bk10, bk11;
        bk00.q[0] = *(const uint4*)(kr0 + chOffK);
        bk00.q[1] = *(const uint4*)(kr0 + chOffK + 8);
        bk01.q[0] = *(const uint4*)(kr0 + 32 + chOffK);
        bk01.q[1] = *(const uint4*)(kr0 + 32 + chOffK + 8);
        bk10.q[0] = *(const uint4*)(kr1 + chOffK);
        bk10.q[1] = *(const uint4*)(kr1 + chOffK + 8);
        bk11.q[0] = *(const uint4*)(kr1 + 32 + chOffK);
        bk11.q[1] = *(const uint4*)(kr1 + 32 + chOffK + 8);

        // ---- energy S = Q^T K (S0: even m, S1: odd m) ----
        v8f S0 = (v8f){}, S1 = (v8f){};
        S0 = wmma_bf16(aq0.v, bk00.v, S0);
        S0 = wmma_bf16(aq1.v, bk01.v, S0);
        S1 = wmma_bf16(aq0.v, bk10.v, S1);
        S1 = wmma_bf16(aq1.v, bk11.v, S1);

        // ---- V^T B-operand fragments from LDS (natural m order) ----
        FragAB bv[4];
        #pragma unroll
        for (int t = 0; t < 4; ++t) {
            const unsigned short* vr = &Vlds[buf][t * 16 + col][0];
            bv[t].q[0] = *(const uint4*)(vr + moff);
            bv[t].q[1] = *(const uint4*)(vr + moff + 8);
        }

        // ---- P = exp2(S) via native v_exp_f32, pack via v_perm_b32 ----
        #pragma unroll
        for (int r = 0; r < 8; ++r) {
            unsigned pk = pk2bf_rn(__builtin_amdgcn_exp2f(S0[r]),
                                   __builtin_amdgcn_exp2f(S1[r]));
            *(unsigned*)&Plds[wave][rbase + r][2 * col] = pk;  // m = 2col, 2col+1
        }
        // same-wave LDS ordering: stores above, loads below
        asm volatile("s_wait_dscnt 0x0" ::: "memory");

        // ---- P as A-operand from LDS ----
        FragAB ap;
        const unsigned short* prow = &Plds[wave][col][0];
        ap.q[0] = *(const uint4*)(prow + kmoff);
        ap.q[1] = *(const uint4*)(prow + kmoff + 8);

        // ---- row normalizer via WMMA against ones, plus out^T += P * V^T ----
        Lacc = wmma_bf16(ap.v, bones.v, Lacc);
        #pragma unroll
        for (int t = 0; t < 4; ++t)
            Oacc[t] = wmma_bf16(ap.v, bv[t].v, Oacc[t]);
    }

    // ---- normalize (1 divide per row), add residual, store fp32 ----
    float rinv[8];
    #pragma unroll
    for (int r = 0; r < 8; ++r) rinv[r] = 1.0f / Lacc[r];

    const float* xb2 = x   + (size_t)b * CCH * NPOS;
    float*       ob  = out + (size_t)b * CCH * NPOS;
    #pragma unroll
    for (int t = 0; t < 4; ++t) {
        int c = t * 16 + col;
        #pragma unroll
        for (int r = 0; r < 8; ++r) {
            unsigned off = (unsigned)c * NPOS + (unsigned)nrow[r];
            ob[off] = Oacc[t][r] * rinv[r] + xb2[off];
        }
    }
}

// ---------------------------------------------------------------------------
extern "C" void kernel_launch(void* const* d_in, const int* in_sizes, int n_in,
                              void* d_out, int out_size, void* d_ws, size_t ws_size,
                              hipStream_t stream) {
    const float* x  = (const float*)d_in[0];
    const float* Wq = (const float*)d_in[1];
    const float* bq = (const float*)d_in[2];
    const float* Wk = (const float*)d_in[3];
    const float* bk = (const float*)d_in[4];
    const float* Wv = (const float*)d_in[5];
    const float* bv = (const float*)d_in[6];
    float* out = (float*)d_out;

    unsigned short* q_pm = (unsigned short*)d_ws;
    unsigned short* k_pm = q_pm + (size_t)BATCH * NPOS * CCH;
    unsigned short* v_cm = k_pm + (size_t)BATCH * NPOS * CCH;

    qkv_proj_kernel<<<(BATCH * NPOS) / 256, 256, 0, stream>>>(
        x, Wq, bq, Wk, bk, Wv, bv, q_pm, k_pm, v_cm);

    // BATCH * 36 blocks * 8 WGs = 576 workgroups; 8 waves each = one 16-row tile/wave
    block_attn_kernel<<<576, 256, 0, stream>>>(x, q_pm, k_pm, v_cm, out);
}
